// Block_58136677318870
// MI455X (gfx1250) — matrix operand order repaired and checked by reading, not compiled, not run on hardware
//
#include <hip/hip_runtime.h>
#include <hip/hip_bf16.h>

typedef __bf16 bf16;
typedef __attribute__((ext_vector_type(16))) __bf16 v16bf;
typedef __attribute__((ext_vector_type(8)))  __bf16 v8bf;
typedef __attribute__((ext_vector_type(8)))  float  v8f;
typedef uint32_t u32x4 __attribute__((ext_vector_type(4)));
typedef uint32_t u32x8 __attribute__((ext_vector_type(8)));

#define DIMN    2048
#define NHEADS  16
#define HDIM    128
#define SEQ     2048
#define BATCH   2
#define HIDDEN  5632
#define MROWS   (BATCH*SEQ)
#define EPSF    1e-6f

__device__ inline v16bf join8(v8bf lo, v8bf hi) {
  return __builtin_shufflevector(lo, hi, 0,1,2,3,4,5,6,7,8,9,10,11,12,13,14,15);
}
__device__ inline v8f zero8() {
  v8f z = {0.f,0.f,0.f,0.f,0.f,0.f,0.f,0.f};
  return z;
}

// ---------------------------------------------------------------------------
// VALU xor-reductions within 16-lane rows via v_permlane16_b32 (no LDS unit).
// Select nibbles: lane i reads lane i^m, same pattern in both wave32 halves.
// ---------------------------------------------------------------------------
__device__ inline float permx(float v, uint32_t lo, uint32_t hi) {
  uint32_t u = __float_as_uint(v);
  uint32_t r = __builtin_amdgcn_permlane16(u, u, lo, hi, false, false);
  return __uint_as_float(r);
}
__device__ inline float redmax16(float v) {
  v = fmaxf(v, permx(v, 0x67452301u, 0xEFCDAB89u));  // xor 1
  v = fmaxf(v, permx(v, 0x54761032u, 0xDCFE98BAu));  // xor 2
  v = fmaxf(v, permx(v, 0x32107654u, 0xBA98FEDCu));  // xor 4
  v = fmaxf(v, permx(v, 0xFEDCBA98u, 0x76543210u));  // xor 8
  return v;
}
__device__ inline float redsum16(float v) {
  v += permx(v, 0x67452301u, 0xEFCDAB89u);
  v += permx(v, 0x54761032u, 0xDCFE98BAu);
  v += permx(v, 0x32107654u, 0xBA98FEDCu);
  v += permx(v, 0xFEDCBA98u, 0x76543210u);
  return v;
}

// ---------------------------------------------------------------------------
// TDM: 2D tile (tile_dim0 = 32 elems contiguous, tile_dim1 = 128 rows) of bf16
// from global [rows][stride0] into contiguous LDS. D# per ISA 8.3-8.6.
// Issued once per wave (EXEC-ignored), tracked by TENSORcnt.
// ---------------------------------------------------------------------------
__device__ inline void tdm_load_tile(uint32_t lds_addr, const bf16* gptr,
                                     uint32_t dim0, uint32_t rows, uint32_t stride0)
{
  uint64_t ga = (uint64_t)(uintptr_t)gptr;
  u32x4 g0;
  g0[0] = 1u;                                            // count=1, user mode
  g0[1] = lds_addr;                                      // LDS byte address
  g0[2] = (uint32_t)ga;                                  // global_addr[31:0]
  g0[3] = (uint32_t)((ga >> 32) & 0x01FFFFFFu) | (2u << 30); // addr[56:32] | type=2
  u32x8 g1;
  g1[0] = (1u << 16);                                    // data_size=2B, mask=0
  g1[1] = (dim0 & 0xFFFFu) << 16;                        // tensor_dim0[15:0]
  g1[2] = (dim0 >> 16) | ((rows & 0xFFFFu) << 16);       // dim0 hi | tensor_dim1 lo
  g1[3] = (rows >> 16) | (32u << 16);                    // dim1 hi | tile_dim0=32
  g1[4] = 128u;                                          // tile_dim1=128, tile_dim2=0
  g1[5] = stride0;                                       // tensor_dim0_stride lo32
  g1[6] = 0u;                                            // stride0 hi | stride1 lo
  g1[7] = 0u;
  u32x4 gz = {0u, 0u, 0u, 0u};
  asm volatile("tensor_load_to_lds %0, %1, %2, %3"
               :: "s"(g0), "s"(g1), "s"(gz), "s"(gz)
               : "memory");
}

// ---------------------------------------------------------------------------
// bf16 GEMM: C[M,N] = A[M,K] x B[N,K]^T, fp32 accum, v_wmma_f32_16x16x32_bf16.
// TDM double-buffered LDS staging: wave 0 issues DMA for tile k+1 while all
// waves run WMMA on tile k. Block 128x128, 8 waves (4Mx2N), wave tile 32x64.
// ---------------------------------------------------------------------------
__global__ __launch_bounds__(256)
void gemm_bf16(const bf16* __restrict__ A, const bf16* __restrict__ Bm,
               float* __restrict__ C, int M, int N, int K)
{
  __shared__ bf16 As[2][128*32];   // [m][k]
  __shared__ bf16 Bs[2][128*32];   // [n][k]
  const int tid  = threadIdx.x;
  const int wave = tid >> 5;
  const int lane = tid & 31;
  const int m0 = blockIdx.y * 128;
  const int n0 = blockIdx.x * 128;
  const int waveM = wave >> 1;
  const int waveN = wave & 1;

  const bf16* Abase = A  + (size_t)m0 * K;
  const bf16* Bbase = Bm + (size_t)n0 * K;

  v8f acc[2][4];
  #pragma unroll
  for (int i = 0; i < 2; i++)
    #pragma unroll
    for (int j = 0; j < 4; j++) acc[i][j] = zero8();

  if (wave == 0) { // prologue: DMA tile 0 into buffer 0
    tdm_load_tile((uint32_t)(uintptr_t)&As[0][0], Abase, (uint32_t)K, 128u, (uint32_t)K);
    tdm_load_tile((uint32_t)(uintptr_t)&Bs[0][0], Bbase, (uint32_t)K, 128u, (uint32_t)K);
  }

  int buf = 0;
  for (int k0 = 0; k0 < K; k0 += 32) {
    const bool has_next = (k0 + 32) < K;
    if (wave == 0) {
      if (has_next) { // prefetch next tile into the other buffer
        tdm_load_tile((uint32_t)(uintptr_t)&As[buf^1][0], Abase + k0 + 32,
                      (uint32_t)(K - k0 - 32), 128u, (uint32_t)K);
        tdm_load_tile((uint32_t)(uintptr_t)&Bs[buf^1][0], Bbase + k0 + 32,
                      (uint32_t)(K - k0 - 32), 128u, (uint32_t)K);
        __builtin_amdgcn_s_wait_tensorcnt(2);  // current tile complete
      } else {
        __builtin_amdgcn_s_wait_tensorcnt(0);
      }
    }
    __syncthreads();

    // Fragments per ISA layout tables
    v16bf afrag[2], bfrag[4];
    const int akb = (lane >= 16) ? 8 : 0;
    const int bkb = (lane >= 16) ? 8 : 0;
    #pragma unroll
    for (int mt = 0; mt < 2; mt++) {
      int row = waveM*32 + mt*16 + (lane & 15);
      v8bf lo = *(const v8bf*)(&As[buf][row*32 + akb]);
      v8bf hi = *(const v8bf*)(&As[buf][row*32 + akb + 16]);
      afrag[mt] = join8(lo, hi);
    }
    #pragma unroll
    for (int nt = 0; nt < 4; nt++) {
      int n = waveN*64 + nt*16 + (lane & 15);
      v8bf lo = *(const v8bf*)(&Bs[buf][n*32 + bkb]);
      v8bf hi = *(const v8bf*)(&Bs[buf][n*32 + bkb + 16]);
      bfrag[nt] = join8(lo, hi);
    }
    #pragma unroll
    for (int mt = 0; mt < 2; mt++)
      #pragma unroll
      for (int nt = 0; nt < 4; nt++)
        acc[mt][nt] = __builtin_amdgcn_wmma_f32_16x16x32_bf16(
            false, afrag[mt], false, bfrag[nt], (short)0, acc[mt][nt], false, false);
    __syncthreads();   // everyone done with buf before DMA overwrites it
    buf ^= 1;
  }

  const int cn    = lane & 15;
  const int rhalf = (lane >> 4) * 8;
  #pragma unroll
  for (int mt = 0; mt < 2; mt++)
    #pragma unroll
    for (int nt = 0; nt < 4; nt++)
      #pragma unroll
      for (int r = 0; r < 8; r++) {
        int row = m0 + waveM*32 + mt*16 + rhalf + r;
        int col = n0 + waveN*64 + nt*16 + cn;
        C[(size_t)row * N + col] = acc[mt][nt][r];
      }
}

// ---------------------------------------------------------------------------
// Flash attention (causal), bf16 WMMA, fp32 online softmax via permlane16.
// ---------------------------------------------------------------------------
#define FBM 128
#define FBN 32

__global__ __launch_bounds__(256)
void flash_attn(const bf16* __restrict__ Q, const bf16* __restrict__ Km,
                const bf16* __restrict__ Vm, float* __restrict__ O)
{
  __shared__ bf16 Qs[FBM*HDIM];   // 32KB  [q][d]
  __shared__ bf16 Ks[FBN*HDIM];   //  8KB  [key][d]
  __shared__ bf16 Vt[HDIM*FBN];   //  8KB  [d][key]
  __shared__ bf16 Ps[8][16*FBN];  //  8KB  per-wave P transpose buffer

  const int tid  = threadIdx.x;
  const int wave = tid >> 5;
  const int lane = tid & 31;
  const int qbase = blockIdx.x * FBM;
  const int bh = blockIdx.y;
  const int b  = bh / NHEADS;
  const int h  = bh % NHEADS;

  const size_t rs = (size_t)NHEADS * HDIM;
  const bf16* Qg = Q  + (size_t)b * SEQ * rs + h * HDIM;
  const bf16* Kg = Km + (size_t)b * SEQ * rs + h * HDIM;
  const bf16* Vg = Vm + (size_t)b * SEQ * rs + h * HDIM;

  {
    int row = tid >> 1;
    int c0  = (tid & 1) * 64;
    const bf16* src = Qg + (size_t)(qbase + row) * rs + c0;
    #pragma unroll
    for (int i = 0; i < 8; i++)
      *(uint4*)(&Qs[row*HDIM + c0 + i*8]) = *(const uint4*)(src + i*8);
  }
  __syncthreads();

  v16bf qfrag[4];
  {
    int row = wave*16 + (lane & 15);
    int kb  = (lane >= 16) ? 8 : 0;
    #pragma unroll
    for (int kk = 0; kk < 4; kk++) {
      v8bf lo = *(const v8bf*)(&Qs[row*HDIM + kk*32 + kb]);
      v8bf hi = *(const v8bf*)(&Qs[row*HDIM + kk*32 + kb + 16]);
      qfrag[kk] = join8(lo, hi);
    }
  }

  v8f oacc[8];
  #pragma unroll
  for (int i = 0; i < 8; i++) oacc[i] = zero8();
  float mrow[8], lrow[8];
  #pragma unroll
  for (int r = 0; r < 8; r++) { mrow[r] = -3.0e38f; lrow[r] = 0.0f; }

  const float scale = 0.08838834764831845f;   // 1/sqrt(128)
  const float LOG2E = 1.4426950408889634f;

  const int jend = qbase + FBM;
  for (int j0 = 0; j0 < jend; j0 += FBN) {
    {
      int key = tid >> 3;
      int c0  = (tid & 7) * 16;
      const bf16* ksrc = Kg + (size_t)(j0 + key) * rs + c0;
      *(uint4*)(&Ks[key*HDIM + c0])     = *(const uint4*)(ksrc);
      *(uint4*)(&Ks[key*HDIM + c0 + 8]) = *(const uint4*)(ksrc + 8);
      const bf16* vsrc = Vg + (size_t)(j0 + key) * rs + c0;
      alignas(16) bf16 tmp[16];
      *(uint4*)(&tmp[0]) = *(const uint4*)(vsrc);
      *(uint4*)(&tmp[8]) = *(const uint4*)(vsrc + 8);
      #pragma unroll
      for (int i = 0; i < 16; i++) Vt[(c0 + i)*FBN + key] = tmp[i];
    }
    __syncthreads();

    v8f sacc[2] = { zero8(), zero8() };
    const int bkb = (lane >= 16) ? 8 : 0;
    #pragma unroll
    for (int nt = 0; nt < 2; nt++) {
      int key = nt*16 + (lane & 15);
      #pragma unroll
      for (int kk = 0; kk < 4; kk++) {
        v8bf lo = *(const v8bf*)(&Ks[key*HDIM + kk*32 + bkb]);
        v8bf hi = *(const v8bf*)(&Ks[key*HDIM + kk*32 + bkb + 16]);
        v16bf bf = join8(lo, hi);
        sacc[nt] = __builtin_amdgcn_wmma_f32_16x16x32_bf16(
            false, qfrag[kk], false, bf, (short)0, sacc[nt], false, false);
      }
    }

    const int qrow0 = qbase + wave*16 + (lane >> 4)*8;
    float pv[2][8];
    #pragma unroll
    for (int r = 0; r < 8; r++) {
      float s0 = sacc[0][r] * scale;
      float s1 = sacc[1][r] * scale;
      int key0 = j0 + (lane & 15);
      int q    = qrow0 + r;
      if (key0 > q)      s0 = -3.0e38f;
      if (key0 + 16 > q) s1 = -3.0e38f;
      float mx = redmax16(fmaxf(s0, s1));
      float mnew = fmaxf(mrow[r], mx);
      float p0 = exp2f((s0 - mnew) * LOG2E);
      float p1 = exp2f((s1 - mnew) * LOG2E);
      pv[0][r] = p0; pv[1][r] = p1;
      float ls = redsum16(p0 + p1);
      float corr = exp2f((mrow[r] - mnew) * LOG2E);
      lrow[r] = lrow[r] * corr + ls;
      mrow[r] = mnew;
      #pragma unroll
      for (int nt = 0; nt < 8; nt++) oacc[nt][r] *= corr;
    }

    #pragma unroll
    for (int nt = 0; nt < 2; nt++)
      #pragma unroll
      for (int r = 0; r < 8; r++)
        Ps[wave][((lane >> 4)*8 + r)*FBN + nt*16 + (lane & 15)] = (bf16)pv[nt][r];
    asm volatile("s_wait_dscnt 0" ::: "memory");

    v16bf pfrag;
    {
      int row = lane & 15;
      int kb  = (lane >= 16) ? 8 : 0;
      v8bf lo = *(const v8bf*)(&Ps[wave][row*FBN + kb]);
      v8bf hi = *(const v8bf*)(&Ps[wave][row*FBN + kb + 16]);
      pfrag = join8(lo, hi);
    }
    #pragma unroll
    for (int nt = 0; nt < 8; nt++) {
      int d = nt*16 + (lane & 15);
      v8bf lo = *(const v8bf*)(&Vt[d*FBN + bkb]);
      v8bf hi = *(const v8bf*)(&Vt[d*FBN + bkb + 16]);
      v16bf vf = join8(lo, hi);
      oacc[nt] = __builtin_amdgcn_wmma_f32_16x16x32_bf16(
          false, pfrag, false, vf, (short)0, oacc[nt], false, false);
    }
    __syncthreads();
  }

  float* Og = O + (size_t)b * SEQ * rs + h * HDIM;
  const int cn = lane & 15;
  #pragma unroll
  for (int r = 0; r < 8; r++) {
    float inv = 1.0f / lrow[r];
    int qrow = qbase + wave*16 + (lane >> 4)*8 + r;
    #pragma unroll
    for (int nt = 0; nt < 8; nt++)
      Og[(size_t)qrow * rs + nt*16 + cn] = oacc[nt][r] * inv;
  }
}

// ---------------------------------------------------------------------------
// Elementwise / layout kernels
// ---------------------------------------------------------------------------
__global__ void cast_f32_bf16(const float* __restrict__ in, bf16* __restrict__ out, size_t n) {
  size_t i = (size_t)blockIdx.x * blockDim.x + threadIdx.x;
  size_t stride = (size_t)gridDim.x * blockDim.x;
  for (; i < n; i += stride) out[i] = (bf16)in[i];
}

// in [R][C] f32 -> out [C][R] bf16 (tiled transpose; R, C multiples of 32)
__global__ __launch_bounds__(256)
void cast_transpose_f32_bf16(const float* __restrict__ in, bf16* __restrict__ out,
                             int R, int Ccols) {
  __shared__ bf16 tile[32][33];
  int c0 = blockIdx.x * 32, r0 = blockIdx.y * 32;
  int tx = threadIdx.x & 31, ty = threadIdx.x >> 5;   // 8 rows per pass
  #pragma unroll
  for (int i = 0; i < 32; i += 8)
    tile[ty + i][tx] = (bf16)in[(size_t)(r0 + ty + i) * Ccols + c0 + tx];
  __syncthreads();
  #pragma unroll
  for (int i = 0; i < 32; i += 8)
    out[(size_t)(c0 + ty + i) * R + r0 + tx] = tile[tx][ty + i];
}

__global__ void add_f32(const float* __restrict__ a, const float* __restrict__ b,
                        float* __restrict__ out, size_t n) {
  size_t i = (size_t)blockIdx.x * blockDim.x + threadIdx.x;
  size_t stride = (size_t)gridDim.x * blockDim.x;
  for (; i < n; i += stride) out[i] = a[i] + b[i];
}

__global__ void rope_cast(const float* __restrict__ qf, const float* __restrict__ kf,
                          const float* __restrict__ cosb, const float* __restrict__ sinb,
                          bf16* __restrict__ qb, bf16* __restrict__ kb) {
  size_t idx = (size_t)blockIdx.x * blockDim.x + threadIdx.x;
  size_t total = (size_t)BATCH * SEQ * NHEADS * (HDIM/2);
  if (idx >= total) return;
  size_t row = idx >> 6;
  int d = (int)(idx & 63);
  int s = (int)((row / NHEADS) % SEQ);
  float c  = cosb[s*64 + d];
  float sn = sinb[s*64 + d];
  size_t off = row * HDIM + 2*(size_t)d;
  float q0 = qf[off], q1 = qf[off+1];
  qb[off]   = (bf16)(q0*c - q1*sn);
  qb[off+1] = (bf16)(q0*sn + q1*c);
  float k0 = kf[off], k1 = kf[off+1];
  kb[off]   = (bf16)(k0*c - k1*sn);
  kb[off+1] = (bf16)(k0*sn + k1*c);
}

__global__ __launch_bounds__(256)
void rmsnorm_k(const float* __restrict__ in, const float* __restrict__ w,
               float* __restrict__ out, bf16* __restrict__ outb) {
  __shared__ float red[8];
  int row = blockIdx.x;
  const float* x = in + (size_t)row * DIMN;
  float ss = 0.f;
  for (int i = threadIdx.x; i < DIMN; i += 256) { float v = x[i]; ss += v*v; }
  ss = redsum16(ss);
  ss += permx(ss, 0, 0) * 0.f;  // keep value
  // cross-half (xor 16) via bpermute-free path: use __shfl_xor once
  ss += __shfl_xor(ss, 16, 32);
  if ((threadIdx.x & 31) == 0) red[threadIdx.x >> 5] = ss;
  __syncthreads();
  if (threadIdx.x == 0) {
    float t = 0.f;
    #pragma unroll
    for (int i = 0; i < 8; i++) t += red[i];
    red[0] = rsqrtf(t / (float)DIMN + EPSF);
  }
  __syncthreads();
  float r = red[0];
  for (int i = threadIdx.x; i < DIMN; i += 256) {
    float v = x[i] * r * w[i];
    out[(size_t)row * DIMN + i] = v;
    if (outb) outb[(size_t)row * DIMN + i] = (bf16)v;
  }
}

__global__ void silu_mul_cast(const float* __restrict__ g, const float* __restrict__ u,
                              bf16* __restrict__ out, size_t n) {
  size_t i = (size_t)blockIdx.x * blockDim.x + threadIdx.x;
  size_t stride = (size_t)gridDim.x * blockDim.x;
  for (; i < n; i += stride) {
    float gv = g[i];
    float s = gv / (1.0f + __expf(-gv));
    out[i] = (bf16)(s * u[i]);
  }
}

// ---------------------------------------------------------------------------
extern "C" void kernel_launch(void* const* d_in, const int* in_sizes, int n_in,
                              void* d_out, int out_size, void* d_ws, size_t ws_size,
                              hipStream_t stream) {
  const float* x     = (const float*)d_in[0];
  const float* wq    = (const float*)d_in[1];
  const float* wk    = (const float*)d_in[2];
  const float* wv    = (const float*)d_in[3];
  const float* wo    = (const float*)d_in[4];
  const float* w_mlp = (const float*)d_in[5];
  const float* v_mlp = (const float*)d_in[6];
  const float* w2    = (const float*)d_in[7];
  const float* n1w   = (const float*)d_in[8];
  const float* n2w   = (const float*)d_in[9];
  const float* fcos  = (const float*)d_in[10];
  const float* fsin  = (const float*)d_in[11];
  float* out = (float*)d_out;

  const size_t n_x  = (size_t)MROWS * DIMN;
  const size_t n_w  = (size_t)DIMN * DIMN;
  const size_t n_wm = (size_t)HIDDEN * DIMN;
  const size_t n_h  = (size_t)MROWS * HIDDEN;

  char* ws = (char*)d_ws;
  size_t cur = 0;
  auto alloc = [&](size_t bytes) -> void* {
    void* p = ws + cur;
    cur = (cur + bytes + 255) & ~(size_t)255;
    return p;
  };
  bf16* wqb = (bf16*)alloc(n_w*2);    // transposed [N][K]
  bf16* wkb = (bf16*)alloc(n_w*2);
  bf16* wvb = (bf16*)alloc(n_w*2);
  bf16* wob = (bf16*)alloc(n_w*2);
  bf16* wmb = (bf16*)alloc(n_wm*2);   // already [N][K]
  bf16* vmb = (bf16*)alloc(n_wm*2);
  bf16* w2b = (bf16*)alloc(n_wm*2);
  bf16* xb  = (bf16*)alloc(n_x*2);
  float* qf = (float*)alloc(n_x*4);
  float* kf = (float*)alloc(n_x*4);
  float* vf = (float*)alloc(n_x*4);
  bf16* qb  = (bf16*)alloc(n_x*2);
  bf16* kb  = (bf16*)alloc(n_x*2);
  bf16* vb  = (bf16*)alloc(n_x*2);
  float* of = (float*)alloc(n_x*4);
  bf16* ob  = (bf16*)alloc(n_x*2);
  float* pf = (float*)alloc(n_x*4);
  float* x2 = (float*)alloc(n_x*4);
  bf16* x2b = (bf16*)alloc(n_x*2);
  float* gatef = (float*)alloc(n_h*4);
  float* upf   = (float*)alloc(n_h*4);
  bf16*  actb  = (bf16*)alloc(n_h*2);
  float* downf = (float*)alloc(n_x*4);

  auto cast = [&](const float* src, bf16* dst, size_t n) {
    int blocks = (int)((n + 255) / 256);
    cast_f32_bf16<<<blocks, 256, 0, stream>>>(src, dst, n);
  };
  auto castT = [&](const float* src, bf16* dst, int R, int Ccols) {
    cast_transpose_f32_bf16<<<dim3(Ccols/32, R/32), 256, 0, stream>>>(src, dst, R, Ccols);
  };

  // 1. weight conversions: [K][N] weights transposed to [N][K] bf16 once
  castT(wq, wqb, DIMN, DIMN); castT(wk, wkb, DIMN, DIMN);
  castT(wv, wvb, DIMN, DIMN); castT(wo, wob, DIMN, DIMN);
  cast(w_mlp, wmb, n_wm); cast(v_mlp, vmb, n_wm); cast(w2, w2b, n_wm);
  cast(x, xb, n_x);

  // 2. QKV projections
  dim3 g_proj(DIMN/128, MROWS/128);
  gemm_bf16<<<g_proj, 256, 0, stream>>>(xb, wqb, qf, MROWS, DIMN, DIMN);
  gemm_bf16<<<g_proj, 256, 0, stream>>>(xb, wkb, kf, MROWS, DIMN, DIMN);
  gemm_bf16<<<g_proj, 256, 0, stream>>>(xb, wvb, vf, MROWS, DIMN, DIMN);

  // 3. RoPE(q,k) + cast; cast v
  {
    size_t pairs = (size_t)BATCH * SEQ * NHEADS * (HDIM/2);
    int blocks = (int)((pairs + 255) / 256);
    rope_cast<<<blocks, 256, 0, stream>>>(qf, kf, fcos, fsin, qb, kb);
  }
  cast(vf, vb, n_x);

  // 4. Flash attention
  flash_attn<<<dim3(SEQ/FBM, BATCH*NHEADS), 256, 0, stream>>>(qb, kb, vb, of);

  // 5. Output projection + residual
  cast(of, ob, n_x);
  gemm_bf16<<<g_proj, 256, 0, stream>>>(ob, wob, pf, MROWS, DIMN, DIMN);
  {
    int blocks = (int)((n_x + 255) / 256);
    add_f32<<<blocks, 256, 0, stream>>>(x, pf, pf, n_x);
  }

  // 6. RMSNorm 1 (fused bf16 copy)
  rmsnorm_k<<<MROWS, 256, 0, stream>>>(pf, n1w, x2, x2b);

  // 7. MLP
  dim3 g_up(HIDDEN/128, MROWS/128);
  gemm_bf16<<<g_up, 256, 0, stream>>>(x2b, wmb, gatef, MROWS, HIDDEN, DIMN);
  gemm_bf16<<<g_up, 256, 0, stream>>>(x2b, vmb, upf,   MROWS, HIDDEN, DIMN);
  {
    int blocks = (int)((n_h + 255) / 256);
    silu_mul_cast<<<blocks, 256, 0, stream>>>(gatef, upf, actb, n_h);
  }
  gemm_bf16<<<g_proj, 256, 0, stream>>>(actb, w2b, downf, MROWS, DIMN, HIDDEN);

  // 8. residual + RMSNorm 2 -> d_out
  {
    int blocks = (int)((n_x + 255) / 256);
    add_f32<<<blocks, 256, 0, stream>>>(x2, downf, downf, n_x);
  }
  rmsnorm_k<<<MROWS, 256, 0, stream>>>(downf, n2w, out, (bf16*)nullptr);
}
